// SSH_46729244180769
// MI455X (gfx1250) — matrix operand order repaired and checked by reading, not compiled, not run on hardware
//
#include <hip/hip_runtime.h>
#include <hip/hip_bf16.h>

typedef __attribute__((ext_vector_type(16))) _Float16 v16h;
typedef __attribute__((ext_vector_type(8)))  float    v8f;

#define BN_EPS 1e-5f

// Padded cin slot per (row,col) LDS pixel: 32 used + 16 pad halves.
// 96B stride keeps 32B alignment for v16h ds loads and spreads banks.
#define CPAD 48

// ------------------------------------------------------------------
// Weight ternarization (BitNet median scale) + packing into the WMMA
// A-fragment layout:  apack[((tap*KC + kc)*mtot + mt)*512 + lane*16 + j]
// with, for cin-within-chunk kk:  g=kk>>4, j = kk&15,
// lane = g*16 + (oc%16)   (contiguous k = 16*g + j, same order as the B reads).
// mtbase/ocbase let two convs share one packed buffer (fused kernels).
// Folds s*gamma/rsqrt(var+eps) and beta-mean*... into alpha/beta.
// ------------------------------------------------------------------
template<int CIN>
__launch_bounds__(256)
__global__ void qpack_kernel(const float* __restrict__ w,
                             const float* __restrict__ gam,
                             const float* __restrict__ bet,
                             const float* __restrict__ rmean,
                             const float* __restrict__ rvar,
                             _Float16* __restrict__ apack,
                             float* __restrict__ alpha,
                             float* __restrict__ betaArr,
                             int mtot, int mtbase, int ocbase) {
  constexpr int N   = CIN * 9;
  constexpr int PER = (N + 255) / 256;
  constexpr int KC  = CIN / 32;
  const int oc  = blockIdx.x;
  const int tid = threadIdx.x;
  const float* wr = w + (size_t)oc * N;

  float    wv[PER];
  unsigned ab[PER];
  bool     ok[PER];
#pragma unroll
  for (int i = 0; i < PER; ++i) {
    int e = tid * PER + i;
    ok[i]   = (e < N);
    float v = ok[i] ? wr[e] : 0.0f;
    wv[i]   = v;
    ab[i]   = __float_as_uint(fabsf(v));
  }

  __shared__ int scnt;
  // Exact median of N (even) values: mean of order stats N/2-1 and N/2,
  // found by binary search on the uint bit pattern of |w| (monotonic >= 0).
  float med[2];
  for (int q = 0; q < 2; ++q) {
    const int rank = N / 2 - 1 + q;
    unsigned lo = 0u, hi = 0x7f800000u;
    while (lo < hi) {
      unsigned mid = (lo + hi) >> 1;
      __syncthreads();
      if (tid == 0) scnt = 0;
      __syncthreads();
      int c = 0;
#pragma unroll
      for (int i = 0; i < PER; ++i)
        if (ok[i] && ab[i] <= mid) ++c;
      if (c) atomicAdd(&scnt, c);
      __syncthreads();
      if (scnt > rank) hi = mid; else lo = mid + 1;   // uniform across block
    }
    med[q] = __uint_as_float(lo);
  }
  const float s = fmaxf(0.5f * (med[0] + med[1]), 1e-5f);

  const int mt = mtbase + (oc >> 4), mrem = oc & 15;
#pragma unroll
  for (int i = 0; i < PER; ++i) {
    if (!ok[i]) continue;
    int e   = tid * PER + i;
    int ci  = e / 9, tap = e % 9;
    int kc  = ci >> 5, kk = ci & 31;
    int g   = kk >> 4;
    int j   = kk & 15;
    int lane = g * 16 + mrem;
    float qv = fminf(1.0f, fmaxf(-1.0f, rintf(wv[i] / s)));  // ternary: exact in f16
    apack[((((size_t)tap * KC + kc) * mtot + mt) * 32 + lane) * 16 + j] = (_Float16)qv;
  }
  if (tid == 0) {
    float inv = gam[oc] * rsqrtf(rvar[oc] + BN_EPS);
    alpha[ocbase + oc]   = s * inv;
    betaArr[ocbase + oc] = bet[oc] - rmean[oc] * inv;
  }
}

// ------------------------------------------------------------------
// Fused implicit-GEMM 3x3 conv pair via v_wmma_f32_16x16x32_f16.
// Block = 128 threads (4 waves) per (batch, output row). One LDS halo
// tile [3][66][CPAD] per 32-cin chunk feeds ALL output channels of up
// to two convs (MT 16-row M-tiles total; first MSPLIT tiles -> f32
// concat slice of d_out, rest -> f16 workspace for the next layer).
// Each wave owns MT/4 M-tiles: A fragments are loaded once per block
// and reused over 4 hoisted B fragments (from LDS).
// ------------------------------------------------------------------
template<int CIN, int MT, int MSPLIT, typename TIN>
__launch_bounds__(128)
__global__ void conv_wmma_kernel(const TIN* __restrict__ x,
                                 const _Float16* __restrict__ apack,
                                 const float* __restrict__ alpha,
                                 const float* __restrict__ beta,
                                 float* __restrict__ outF,
                                 _Float16* __restrict__ outH,
                                 int coff) {
  constexpr int KC  = CIN / 32;
  constexpr int MTL = MT / 4;                   // M-tiles per wave
  __shared__ __align__(32) _Float16 lds[3 * 66 * CPAD];

  const int bh   = blockIdx.x;
  const int b    = bh >> 6, h = bh & 63;
  const int tid  = threadIdx.x;
  const int lane = tid & 31, wvid = tid >> 5;
  const int nn   = lane & 15, gg = lane >> 4;

  v8f acc[MTL][4] = {};

  for (int kc = 0; kc < KC; ++kc) {
    __syncthreads();  // previous chunk's fragment reads done
    // Stage [3 rows][66 cols][32 cin] f16 halo tile (cin innermost, padded);
    // consecutive tid -> consecutive w (coalesced global reads).
    for (int idx = tid; idx < 3 * 66 * 32; idx += 128) {
      int c  = idx % 66;
      int t2 = idx / 66;
      int ci = t2 & 31, r = t2 >> 5;
      int hh = h + r - 1, ww = c - 1;
      float v = 0.0f;
      if (hh >= 0 && hh < 64 && ww >= 0 && ww < 64)
        v = (float)x[(((size_t)b * CIN + kc * 32 + ci) * 64 + hh) * 64 + ww];
      lds[(r * 66 + c) * CPAD + ci] = (_Float16)v;
    }
    __syncthreads();

#pragma unroll
    for (int tap = 0; tap < 9; ++tap) {
      const int dh = tap / 3, dw = tap % 3;
      // Hoist all 4 N-tile B fragments: b[j] = B[k = gg*16+j][n = nn].
      v16h bf[4];
#pragma unroll
      for (int nt = 0; nt < 4; ++nt)
        bf[nt] = *(const v16h*)&lds[(dh * 66 + nt * 16 + nn + dw) * CPAD + gg * 16];
#pragma unroll
      for (int ml = 0; ml < MTL; ++ml) {
        const int mtg = wvid * MTL + ml;
        v16h af = *(const v16h*)(apack +
                    ((((size_t)tap * KC + kc) * MT + mtg) * 32 + lane) * 16);
#pragma unroll
        for (int nt = 0; nt < 4; ++nt)
          acc[ml][nt] = __builtin_amdgcn_wmma_f32_16x16x32_f16(
              false, af, false, bf[nt], (short)0, acc[ml][nt], false, false);
      }
    }
  }

  // D layout: VGPR r -> M = (lane/16)*8 + r, N = lane%16.
#pragma unroll
  for (int ml = 0; ml < MTL; ++ml) {
    const int mtg = wvid * MTL + ml;
#pragma unroll
    for (int nt = 0; nt < 4; ++nt) {
      const int wcol = nt * 16 + nn;
#pragma unroll
      for (int r = 0; r < 8; ++r) {
        int oc  = mtg * 16 + gg * 8 + r;
        float y = acc[ml][nt][r] * alpha[oc] + beta[oc];
        y = y / (1.0f + __expf(-y));  // SiLU
        if (oc < MSPLIT * 16)
          outF[(((size_t)b * 256 + coff + oc) * 64 + h) * 64 + wcol] = y;
        else
          outH[(((size_t)b * 64 + (oc - MSPLIT * 16)) * 64 + h) * 64 + wcol] =
              (_Float16)y;
      }
    }
  }
}

// ------------------------------------------------------------------
extern "C" void kernel_launch(void* const* d_in, const int* in_sizes, int n_in,
                              void* d_out, int out_size, void* d_ws, size_t ws_size,
                              hipStream_t stream) {
  (void)in_sizes; (void)n_in; (void)out_size; (void)ws_size;
  const float* x   = (const float*)d_in[0];
  const float* w3  = (const float*)d_in[1];
  const float* g3  = (const float*)d_in[2];
  const float* b3  = (const float*)d_in[3];
  const float* m3  = (const float*)d_in[4];
  const float* v3  = (const float*)d_in[5];
  const float* wS  = (const float*)d_in[6];
  const float* gS  = (const float*)d_in[7];
  const float* bS  = (const float*)d_in[8];
  const float* mS  = (const float*)d_in[9];
  const float* vS  = (const float*)d_in[10];
  const float* w5  = (const float*)d_in[11];
  const float* g5  = (const float*)d_in[12];
  const float* b5  = (const float*)d_in[13];
  const float* m5  = (const float*)d_in[14];
  const float* v5  = (const float*)d_in[15];
  const float* w7a = (const float*)d_in[16];
  const float* g7a = (const float*)d_in[17];
  const float* b7a = (const float*)d_in[18];
  const float* m7a = (const float*)d_in[19];
  const float* v7a = (const float*)d_in[20];
  const float* w7b = (const float*)d_in[21];
  const float* g7b = (const float*)d_in[22];
  const float* b7b = (const float*)d_in[23];
  const float* m7b = (const float*)d_in[24];
  const float* v7b = (const float*)d_in[25];

  char* ws = (char*)d_ws;
  size_t off = 0;
  auto take = [&](size_t bytes) -> void* {
    void* p = (void*)(ws + off);
    off += (bytes + 255) & ~(size_t)255;
    return p;
  };
  _Float16* stem = (_Float16*)take((size_t)32 * 64 * 64 * 64 * 2);   // 16.8 MB
  _Float16* tbuf = (_Float16*)take((size_t)32 * 64 * 64 * 64 * 2);   // 16.8 MB
  _Float16* apA  = (_Float16*)take((size_t)9 * 8 * 12 * 512 * 2);    // conv3+convS
  _Float16* apB  = (_Float16*)take((size_t)9 * 2 * 8  * 512 * 2);    // conv5+conv7a
  _Float16* apC  = (_Float16*)take((size_t)9 * 2 * 4  * 512 * 2);    // conv7b
  float* alA = (float*)take(192 * 4); float* beA = (float*)take(192 * 4);
  float* alB = (float*)take(128 * 4); float* beB = (float*)take(128 * 4);
  float* alC = (float*)take(64 * 4);  float* beC = (float*)take(64 * 4);

  // --- ternarize + pack weights (concatenated per fused kernel), fold BN ---
  qpack_kernel<256><<<128, 256, 0, stream>>>(w3,  g3,  b3,  m3,  v3,  apA, alA, beA, 12, 0, 0);
  qpack_kernel<256><<<64,  256, 0, stream>>>(wS,  gS,  bS,  mS,  vS,  apA, alA, beA, 12, 8, 128);
  qpack_kernel<64> <<<64,  256, 0, stream>>>(w5,  g5,  b5,  m5,  v5,  apB, alB, beB, 8, 0, 0);
  qpack_kernel<64> <<<64,  256, 0, stream>>>(w7a, g7a, b7a, m7a, v7a, apB, alB, beB, 8, 4, 64);
  qpack_kernel<64> <<<64,  256, 0, stream>>>(w7b, g7b, b7b, m7b, v7b, apC, alC, beC, 4, 0, 0);

  // --- fused convs: one block per (batch,row) = 2048 blocks, 4 waves each ---
  float* out = (float*)d_out;
  // A: x -> c3 = silu(bn(conv3)) into out[0:128)  AND  stem = silu(bn(convS))
  conv_wmma_kernel<256, 12, 8, float><<<2048, 128, 0, stream>>>(
      x, apA, alA, beA, out, stem, 0);
  // B: stem -> c5 into out[128:192)  AND  t = silu(bn(conv7a))
  conv_wmma_kernel<64, 8, 4, _Float16><<<2048, 128, 0, stream>>>(
      stem, apB, alB, beB, out, tbuf, 128);
  // C: t -> c7 into out[192:256)
  conv_wmma_kernel<64, 4, 4, _Float16><<<2048, 128, 0, stream>>>(
      tbuf, apC, alC, beC, out, nullptr, 192);
}
